// ASENet_V2_38208029066077
// MI455X (gfx1250) — compile-verified, matrix operand order checked by reading, not run on hardware
//
#include <hip/hip_runtime.h>
#include <math.h>
#include <stdint.h>

// ---------------------------------------------------------------------------
// Types for CDNA5 WMMA (wave32, 16x16x32 bf16 -> f32)
// ---------------------------------------------------------------------------
typedef __bf16 bf16_t;
typedef bf16_t v16bf __attribute__((ext_vector_type(16)));
typedef float  v8f   __attribute__((ext_vector_type(8)));

union FragAB { uint32_t u[8]; v16bf v; };
union Bf16U  { unsigned short s; bf16_t b; };

__device__ __forceinline__ bf16_t f2bf(float f) {
    union { float f; uint32_t u; } v; v.f = f;
    uint32_t u = v.u;
    uint32_t r = (u + 0x7FFFu + ((u >> 16) & 1u)) >> 16;   // RNE
    Bf16U o; o.s = (unsigned short)r; return o.b;
}
__device__ __forceinline__ float bf2f(bf16_t b) {
    Bf16U i; i.b = b;
    union { float f; uint32_t u; } o; o.u = ((uint32_t)i.s) << 16; return o.f;
}

__device__ __forceinline__ float wave_reduce_sum(float s) {
    #pragma unroll
    for (int off = 16; off > 0; off >>= 1) s += __shfl_xor(s, off, 32);
    return s;
}

// Raw LDS byte offset (addrspace(3)) for async-to-LDS instructions.
__device__ __forceinline__ uint32_t lds_off(void* p) {
    return (uint32_t)(uintptr_t)(__attribute__((address_space(3))) char*)p;
}

// One wave-wide async 16B/lane copy: global -> LDS (ASYNCcnt-tracked).
__device__ __forceinline__ void async_copy16(uint32_t lds_byte_off, const void* gp) {
    asm volatile("global_load_async_to_lds_b128 %0, %1, off"
                 :: "v"(lds_byte_off), "v"((unsigned long long)(uintptr_t)gp)
                 : "memory");
}

// ---------------------------------------------------------------------------
// Problem constants
// ---------------------------------------------------------------------------
#define BSZ   512
#define CIN   1024
#define CMID  512
#define HW2   196          // 14*14
#define MBIG  (BSZ*HW2)    // 100352
#define EMB   1024

// ---------------------------------------------------------------------------
// Pre-pass: transpose+convert x [B, CIN, 196] f32 -> xT [B*196, CIN] bf16
// ---------------------------------------------------------------------------
__global__ __launch_bounds__(256) void xpose_kernel(const float* __restrict__ x,
                                                    bf16_t* __restrict__ xT) {
    __shared__ float t[32][33];
    const int hw0 = blockIdx.x * 32;
    const int c0  = blockIdx.y * 32;
    const int b   = blockIdx.z;
    const int tx  = threadIdx.x & 31;
    const int ty  = threadIdx.x >> 5;          // 0..7
    const float* xb = x + (size_t)b * CIN * HW2;
    #pragma unroll
    for (int i = 0; i < 4; ++i) {
        int cc = c0 + ty + i * 8;
        int hw = hw0 + tx;
        t[ty + i * 8][tx] = (hw < HW2) ? xb[(size_t)cc * HW2 + hw] : 0.0f;
    }
    __syncthreads();
    bf16_t* ob = xT + (size_t)b * HW2 * CIN;
    #pragma unroll
    for (int i = 0; i < 4; ++i) {
        int hw = hw0 + ty + i * 8;
        int cc = c0 + tx;
        if (hw < HW2) ob[(size_t)hw * CIN + cc] = f2bf(t[tx][ty + i * 8]);
    }
}

// f32 -> bf16 plain convert (weights)
__global__ __launch_bounds__(256) void cvt_bf16_kernel(const float* __restrict__ in,
                                                       bf16_t* __restrict__ out, int n) {
    int i = blockIdx.x * 256 + threadIdx.x;
    if (i < n) out[i] = f2bf(in[i]);
}

// Fold BN (inference) + conv bias into per-channel scale/shift.
__global__ __launch_bounds__(256) void bnfold_kernel(const float* __restrict__ bc,
                                                     const float* __restrict__ gamma,
                                                     const float* __restrict__ beta,
                                                     const float* __restrict__ mean,
                                                     const float* __restrict__ var,
                                                     float* __restrict__ scale,
                                                     float* __restrict__ shift) {
    int i = blockIdx.x * 256 + threadIdx.x;
    if (i < CMID) {
        float sc = gamma[i] * rsqrtf(var[i] + 1e-5f);
        scale[i] = sc;
        shift[i] = (bc[i] - mean[i]) * sc + beta[i];
    }
}

// Per-attribute MLPs (only 8 distinct attributes).
__global__ __launch_bounds__(256) void attr_kernel(const float* __restrict__ emb,
                                                   const float* __restrict__ Wt1,
                                                   const float* __restrict__ bt1,
                                                   const float* __restrict__ Wt2,
                                                   const float* __restrict__ bt2,
                                                   float* __restrict__ attrT,
                                                   float* __restrict__ attr2T) {
    int idx = blockIdx.x * 256 + threadIdx.x;      // 0 .. 8191
    int which = idx >> 12;                          // 0: Wt1, 1: Wt2
    int r = idx & 4095;
    int a = r >> 9, n = r & 511;
    const float* W = which ? Wt2 : Wt1;
    const float* bias = which ? bt2 : bt1;
    const float* e = emb + (size_t)a * 512;
    const float* w = W + (size_t)n * 512;
    float s = bias[n];
    for (int k = 0; k < 512; ++k) s += e[k] * w[k];
    if (which) attr2T[a * 512 + n] = s > 0.0f ? s : 0.0f;
    else       attrT[a * 512 + n]  = tanhf(s);
}

// ---------------------------------------------------------------------------
// Generic WMMA GEMM:  D[m,n] = epilogue( sum_k A[m,k]*B[n,k] )
//   A: [M,K] bf16 row-major, B: [N,K] bf16 row-major.
//   Block tile 128x128, K-step 32; 256 threads = 8 waves; wave tile 64x32.
//   Double-buffered LDS fed by GLOBAL_LOAD_ASYNC_TO_LDS_B128 (ASYNCcnt).
//   EPI 0: bf16 tanh(x*p0[n]+p1[n])   (conv + BN + tanh)
//   EPI 1: bf16 relu(x+p0[n])
//   EPI 2: bf16 sigmoid(x+p0[n])
//   EPI 3: f32  x+p0[n]
// ---------------------------------------------------------------------------
#define BM 128
#define BN 128
#define BK 32

template <int EPI>
__global__ __launch_bounds__(256) void wmma_gemm_kernel(
    const bf16_t* __restrict__ A, const bf16_t* __restrict__ B,
    const float* __restrict__ p0, const float* __restrict__ p1,
    bf16_t* __restrict__ Dbf, float* __restrict__ Df,
    int M, int N, int K) {
    __shared__ __align__(16) bf16_t lA[2][BM][BK];
    __shared__ __align__(16) bf16_t lB[2][BN][BK];

    const int tid   = threadIdx.x;
    const int ntb   = N / BN;
    const int mBase = (blockIdx.x / ntb) * BM;
    const int nBase = (blockIdx.x % ntb) * BN;
    const int w     = tid >> 5;
    const int lane  = tid & 31;
    const int lm    = lane & 15;
    const int lh    = lane >> 4;
    const int wm    = (w & 1) * 64;
    const int wn    = (w >> 1) * 32;

    // Staging assignment: thread -> rows r0 and r0+64, 16B chunk q0 (elements).
    const int r0 = tid >> 2;            // 0..63
    const int q0 = (tid & 3) * 8;       // 0,8,16,24

    const uint32_t ldsA0 = lds_off(&lA[0][0][0]) + (uint32_t)(r0 * BK + q0) * 2u;
    const uint32_t ldsB0 = lds_off(&lB[0][0][0]) + (uint32_t)(r0 * BK + q0) * 2u;
    const uint32_t bufStride = (uint32_t)(BM * BK * 2);   // 8192 B
    const uint32_t rowStride = (uint32_t)(64 * BK * 2);   // rows +64 -> 4096 B

    const bf16_t* Abase = A + (size_t)(mBase + r0) * K + q0;
    const bf16_t* Bbase = B + (size_t)(nBase + r0) * K + q0;
    const size_t  row64 = (size_t)64 * K;

    v8f acc[4][2];
    #pragma unroll
    for (int i = 0; i < 4; ++i)
        #pragma unroll
        for (int j = 0; j < 2; ++j)
            #pragma unroll
            for (int r = 0; r < 8; ++r) acc[i][j][r] = 0.0f;

    // Issue one stage (4 async wave-instructions: 2 A rows + 2 B rows).
    auto issue = [&](int kk, int buf) {
        const uint32_t off = buf ? bufStride : 0u;
        async_copy16(ldsA0 + off,             Abase + kk);
        async_copy16(ldsA0 + off + rowStride, Abase + kk + row64);
        async_copy16(ldsB0 + off,             Bbase + kk);
        async_copy16(ldsB0 + off + rowStride, Bbase + kk + row64);
    };

    issue(0, 0);
    int buf = 0;
    for (int kk = 0; kk < K; kk += BK, buf ^= 1) {
        if (kk + BK < K) {
            __syncthreads();                 // all waves done reading buf^1
            issue(kk + BK, buf ^ 1);
            // deep prefetch (2 stages ahead) to cover HBM/L2 latency
            if (kk + 2 * BK < K) {
                __builtin_prefetch(Abase + kk + 2 * BK, 0, 1);
                __builtin_prefetch(Bbase + kk + 2 * BK, 0, 1);
            }
            asm volatile("s_wait_asynccnt 0x4" ::: "memory");  // prev stage done
        } else {
            asm volatile("s_wait_asynccnt 0x0" ::: "memory");
        }
        __syncthreads();                     // prev stage done across all waves

        // Fragment loads per ISA 16-bit A 16x32 layout:
        //   lane L (L&15 = row), VGPR v holds K pair at
        //   kv = (v>>2)*16 + (L>>4)*8 + (v&3)*2
        FragAB afr[4], bfr[2];
        #pragma unroll
        for (int ms = 0; ms < 4; ++ms) {
            const bf16_t* ap = &lA[buf][wm + ms * 16 + lm][0];
            #pragma unroll
            for (int v = 0; v < 8; ++v) {
                int kv = ((v >> 2) << 4) + (lh << 3) + ((v & 3) << 1);
                afr[ms].u[v] = *(const uint32_t*)(ap + kv);
            }
        }
        #pragma unroll
        for (int ns = 0; ns < 2; ++ns) {
            const bf16_t* bp = &lB[buf][wn + ns * 16 + lm][0];
            #pragma unroll
            for (int v = 0; v < 8; ++v) {
                int kv = ((v >> 2) << 4) + (lh << 3) + ((v & 3) << 1);
                bfr[ns].u[v] = *(const uint32_t*)(bp + kv);
            }
        }
        #pragma unroll
        for (int ms = 0; ms < 4; ++ms)
            #pragma unroll
            for (int ns = 0; ns < 2; ++ns)
                acc[ms][ns] = __builtin_amdgcn_wmma_f32_16x16x32_bf16(
                    false, afr[ms].v, false, bfr[ns].v,
                    (short)0, acc[ms][ns], false, false);
    }

    // Epilogue. C/D layout: VGPR r, lanes 0-15 -> M=r, lanes 16-31 -> M=8+r.
    #pragma unroll
    for (int ms = 0; ms < 4; ++ms)
        #pragma unroll
        for (int ns = 0; ns < 2; ++ns) {
            const int n = nBase + wn + ns * 16 + lm;
            const float q = p0[n];
            #pragma unroll
            for (int r = 0; r < 8; ++r) {
                const int m = mBase + wm + ms * 16 + (lh << 3) + r;
                float xv = acc[ms][ns][r];
                if (EPI == 0) {
                    Dbf[(size_t)m * N + n] = f2bf(tanhf(xv * q + p1[n]));
                } else if (EPI == 1) {
                    float y = xv + q;
                    Dbf[(size_t)m * N + n] = f2bf(y > 0.0f ? y : 0.0f);
                } else if (EPI == 2) {
                    Dbf[(size_t)m * N + n] = f2bf(1.0f / (1.0f + __expf(-(xv + q))));
                } else {
                    Df[(size_t)m * N + n] = xv + q;
                }
            }
        }
}

// ---------------------------------------------------------------------------
// attmap[m] = (attr[b] . img[m,:]) / sqrt(512). One wave per row.
// ---------------------------------------------------------------------------
__global__ __launch_bounds__(256) void attmap_kernel(const bf16_t* __restrict__ img,
                                                     const float* __restrict__ attrT,
                                                     const int* __restrict__ c,
                                                     float* __restrict__ attraw) {
    const int w = threadIdx.x >> 5, lane = threadIdx.x & 31;
    const int m = blockIdx.x * 8 + w;
    const int b = m / HW2;
    const float* av = attrT + (size_t)c[b] * 512;
    const bf16_t* ir = img + (size_t)m * CMID;
    float s = 0.0f;
    for (int j = lane; j < CMID; j += 32) s += bf2f(ir[j]) * av[j];
    s = wave_reduce_sum(s);
    if (lane == 0) attraw[m] = s * 0.04419417382415922f;   // 1/sqrt(512)
}

// Spatial softmax over the 196 positions of each batch row.
__global__ __launch_bounds__(256) void softmax_kernel(const float* __restrict__ raw,
                                                      float* __restrict__ outw) {
    __shared__ float red[8];
    const int b = blockIdx.x, t = threadIdx.x;
    const int w = t >> 5, lane = t & 31;
    float v = (t < HW2) ? raw[b * HW2 + t] : -1e30f;
    float mx = v;
    #pragma unroll
    for (int off = 16; off > 0; off >>= 1) mx = fmaxf(mx, __shfl_xor(mx, off, 32));
    if (lane == 0) red[w] = mx;
    __syncthreads();
    if (t == 0) {
        float m2 = red[0];
        for (int i = 1; i < 8; ++i) m2 = fmaxf(m2, red[i]);
        red[0] = m2;
    }
    __syncthreads();
    mx = red[0];
    __syncthreads();
    float e = (t < HW2) ? __expf(v - mx) : 0.0f;
    float s = wave_reduce_sum(e);
    if (lane == 0) red[w] = s;
    __syncthreads();
    if (t == 0) {
        float s2 = 0.0f;
        for (int i = 0; i < 8; ++i) s2 += red[i];
        red[0] = s2;
    }
    __syncthreads();
    if (t < HW2) outw[b * HW2 + t] = e / red[0];
}

// feat[b,c] = sum_hw x[b,c,hw] * attw[b,hw]. One block per batch row.
__global__ __launch_bounds__(256) void feat_kernel(const float* __restrict__ x,
                                                   const float* __restrict__ attw,
                                                   float* __restrict__ feat) {
    __shared__ float sa[HW2];
    const int b = blockIdx.x, t = threadIdx.x;
    const int w = t >> 5, lane = t & 31;
    if (t < HW2) sa[t] = attw[b * HW2 + t];
    __syncthreads();
    for (int cc = w; cc < CIN; cc += 8) {
        const float* xr = x + ((size_t)b * CIN + cc) * HW2;
        float s = 0.0f;
        for (int j = lane; j < HW2; j += 32) s += xr[j] * sa[j];
        s = wave_reduce_sum(s);
        if (lane == 0) feat[(size_t)b * CIN + cc] = s;
    }
}

// ia[m, 0:1024] = feat[m], ia[m, 1024:1536] = attr2T[c[m]]  (bf16)
__global__ __launch_bounds__(256) void iabuild_kernel(const float* __restrict__ feat,
                                                      const float* __restrict__ attr2T,
                                                      const int* __restrict__ c,
                                                      bf16_t* __restrict__ ia) {
    int i = blockIdx.x * 256 + threadIdx.x;        // < 512*1536
    int m = i / 1536, k = i - m * 1536;
    float v = (k < 1024) ? feat[(size_t)m * 1024 + k]
                         : attr2T[(size_t)c[m] * 512 + (k - 1024)];
    ia[i] = f2bf(v);
}

// g[m,k] = feat[m,k] * mask[m,k]  (bf16)
__global__ __launch_bounds__(256) void gbuild_kernel(const float* __restrict__ feat,
                                                     const bf16_t* __restrict__ mask,
                                                     bf16_t* __restrict__ g) {
    int i = blockIdx.x * 256 + threadIdx.x;        // < 512*1024
    g[i] = f2bf(feat[i] * bf2f(mask[i]));
}

// Row-wise L2 normalization: out = raw / ||raw||_2
__global__ __launch_bounds__(256) void l2norm_kernel(const float* __restrict__ raw,
                                                     float* __restrict__ out) {
    __shared__ float red[8];
    const int b = blockIdx.x, t = threadIdx.x;
    const int w = t >> 5, lane = t & 31;
    float s = 0.0f;
    for (int j = t; j < EMB; j += 256) {
        float v = raw[(size_t)b * EMB + j];
        s += v * v;
    }
    s = wave_reduce_sum(s);
    if (lane == 0) red[w] = s;
    __syncthreads();
    if (t == 0) {
        float s2 = 0.0f;
        for (int i = 0; i < 8; ++i) s2 += red[i];
        red[0] = s2;
    }
    __syncthreads();
    const float inv = rsqrtf(red[0]);
    for (int j = t; j < EMB; j += 256)
        out[(size_t)b * EMB + j] = raw[(size_t)b * EMB + j] * inv;
}

// ---------------------------------------------------------------------------
// Host launch
// ---------------------------------------------------------------------------
extern "C" void kernel_launch(void* const* d_in, const int* in_sizes, int n_in,
                              void* d_out, int out_size, void* d_ws, size_t ws_size,
                              hipStream_t stream) {
    const float* x        = (const float*)d_in[0];
    const int*   c        = (const int*)d_in[1];
    const float* attr_emb = (const float*)d_in[2];
    const float* Wt1      = (const float*)d_in[3];
    const float* bt1      = (const float*)d_in[4];
    const float* Wc       = (const float*)d_in[5];
    const float* bc       = (const float*)d_in[6];
    const float* bn_g     = (const float*)d_in[7];
    const float* bn_b     = (const float*)d_in[8];
    const float* bn_m     = (const float*)d_in[9];
    const float* bn_v     = (const float*)d_in[10];
    const float* Wt2      = (const float*)d_in[11];
    const float* bt2      = (const float*)d_in[12];
    const float* W1       = (const float*)d_in[13];
    const float* b1       = (const float*)d_in[14];
    const float* W2       = (const float*)d_in[15];
    const float* b2       = (const float*)d_in[16];
    const float* Wf       = (const float*)d_in[17];
    const float* bfv      = (const float*)d_in[18];
    float* out = (float*)d_out;

    char* ws = (char*)d_ws;
    size_t cur = 0;
    auto take = [&](size_t bytes) -> void* {
        void* p = ws + cur;
        cur = (cur + bytes + 255) & ~(size_t)255;
        return p;
    };
    bf16_t* xT     = (bf16_t*)take((size_t)MBIG * CIN * 2);
    bf16_t* Wcbf   = (bf16_t*)take((size_t)CMID * CIN * 2);
    bf16_t* imgbf  = (bf16_t*)take((size_t)MBIG * CMID * 2);
    bf16_t* W1bf   = (bf16_t*)take((size_t)512 * 1536 * 2);
    bf16_t* W2bf   = (bf16_t*)take((size_t)1024 * 512 * 2);
    bf16_t* Wfbf   = (bf16_t*)take((size_t)1024 * 1024 * 2);
    float*  bnsc   = (float*)take(512 * 4);
    float*  bnsh   = (float*)take(512 * 4);
    float*  attrT  = (float*)take(8 * 512 * 4);
    float*  attr2T = (float*)take(8 * 512 * 4);
    float*  attraw = (float*)take((size_t)BSZ * HW2 * 4);
    float*  attw   = (float*)take((size_t)BSZ * HW2 * 4);
    float*  feat   = (float*)take((size_t)BSZ * CIN * 4);
    bf16_t* iabf   = (bf16_t*)take((size_t)512 * 1536 * 2);
    bf16_t* hbf    = (bf16_t*)take((size_t)512 * 512 * 2);
    bf16_t* maskbf = (bf16_t*)take((size_t)512 * 1024 * 2);
    bf16_t* gbf    = (bf16_t*)take((size_t)512 * 1024 * 2);
    float*  outraw = (float*)take((size_t)512 * 1024 * 4);

    // --- pre-passes ---
    xpose_kernel<<<dim3(7, 32, BSZ), 256, 0, stream>>>(x, xT);
    cvt_bf16_kernel<<<(CMID * CIN + 255) / 256, 256, 0, stream>>>(Wc, Wcbf, CMID * CIN);
    cvt_bf16_kernel<<<(512 * 1536 + 255) / 256, 256, 0, stream>>>(W1, W1bf, 512 * 1536);
    cvt_bf16_kernel<<<(1024 * 512 + 255) / 256, 256, 0, stream>>>(W2, W2bf, 1024 * 512);
    cvt_bf16_kernel<<<(1024 * 1024 + 255) / 256, 256, 0, stream>>>(Wf, Wfbf, 1024 * 1024);
    bnfold_kernel<<<2, 256, 0, stream>>>(bc, bn_g, bn_b, bn_m, bn_v, bnsc, bnsh);
    attr_kernel<<<32, 256, 0, stream>>>(attr_emb, Wt1, bt1, Wt2, bt2, attrT, attr2T);

    // --- conv1 + BN + tanh: [100352,1024] x [512,1024]^T -> img bf16 ---
    wmma_gemm_kernel<0><<<(MBIG / BM) * (CMID / BN), 256, 0, stream>>>(
        xT, Wcbf, bnsc, bnsh, imgbf, nullptr, MBIG, CMID, CIN);

    // --- attention map + softmax + attended feature ---
    attmap_kernel<<<MBIG / 8, 256, 0, stream>>>(imgbf, attrT, c, attraw);
    softmax_kernel<<<BSZ, 256, 0, stream>>>(attraw, attw);
    feat_kernel<<<BSZ, 256, 0, stream>>>(x, attw, feat);

    // --- gated fusion MLP head (all WMMA) ---
    iabuild_kernel<<<(512 * 1536 + 255) / 256, 256, 0, stream>>>(feat, attr2T, c, iabf);
    wmma_gemm_kernel<1><<<(512 / BM) * (512 / BN), 256, 0, stream>>>(
        iabf, W1bf, b1, nullptr, hbf, nullptr, 512, 512, 1536);
    wmma_gemm_kernel<2><<<(512 / BM) * (1024 / BN), 256, 0, stream>>>(
        hbf, W2bf, b2, nullptr, maskbf, nullptr, 512, 1024, 512);
    gbuild_kernel<<<(512 * 1024 + 255) / 256, 256, 0, stream>>>(feat, maskbf, gbf);
    wmma_gemm_kernel<3><<<(512 / BM) * (1024 / BN), 256, 0, stream>>>(
        gbf, Wfbf, bfv, nullptr, nullptr, outraw, 512, 1024, 1024);

    // --- row-wise l2 normalization -> d_out ---
    l2norm_kernel<<<BSZ, 256, 0, stream>>>(outraw, out);
}
